// AGCRN_34282428956827
// MI455X (gfx1250) — matrix operand order repaired and checked by reading, not compiled, not run on hardware
//
#include <hip/hip_runtime.h>

// ---- problem constants ----
#define NN   2000   // nodes
#define NP   2048   // padded nodes
#define BB   32     // batch
#define HH   64     // hidden
#define DD   10     // embed
#define TIN  24
#define TOUT 12
#define CP0  96     // padded channels layer0 (real 65)
#define KD0  (2*CP0)   // 192
#define CP1  128    // layer1 channels (real 128)
#define KD1  (2*CP1)   // 256
#define BC0  (BB*CP0)  // 3072 columns
#define BC1  (BB*CP1)  // 4096 columns

typedef __attribute__((ext_vector_type(16))) __bf16 v16bf;
typedef __attribute__((ext_vector_type(8)))  float  v8f;
typedef unsigned int u32x4 __attribute__((ext_vector_type(4)));
typedef int          i32x4 __attribute__((ext_vector_type(4)));
typedef int          i32x8 __attribute__((ext_vector_type(8)));

__device__ __forceinline__ unsigned short f2bf(float f) {
  union { float f; unsigned u; } v; v.f = f;
  unsigned r = v.u + 0x7FFFu + ((v.u >> 16) & 1u);
  return (unsigned short)(r >> 16);
}
__device__ __forceinline__ float bf2f(unsigned short h) {
  union { unsigned u; float f; } v; v.u = ((unsigned)h) << 16;
  return v.f;
}

// ---------------- zero fill ----------------
__global__ void k_zero(unsigned int* __restrict__ p, long long nwords) {
  long long i = (long long)blockIdx.x * blockDim.x + threadIdx.x;
  long long st = (long long)gridDim.x * blockDim.x;
  for (; i < nwords; i += st) p[i] = 0u;
}

// ---------------- adjacency: A = softmax(relu(E E^T)) -> bf16 (NP x NP) ----------------
__global__ void k_adj(const float* __restrict__ E, unsigned short* __restrict__ Abf) {
  __shared__ float row[NP];
  __shared__ float red[256];
  __shared__ float en[DD];
  int n = blockIdx.x;            // < NN
  int tid = threadIdx.x;
  if (tid < DD) en[tid] = E[n * DD + tid];
  __syncthreads();
  float mx = 0.0f;               // relu >= 0
  for (int m = tid; m < NN; m += 256) {
    float s = 0.f;
    #pragma unroll
    for (int d = 0; d < DD; ++d) s += en[d] * E[m * DD + d];
    s = s > 0.f ? s : 0.f;
    row[m] = s;
    mx = fmaxf(mx, s);
  }
  red[tid] = mx; __syncthreads();
  for (int o = 128; o > 0; o >>= 1) { if (tid < o) red[tid] = fmaxf(red[tid], red[tid + o]); __syncthreads(); }
  float rmax = red[0]; __syncthreads();
  float sum = 0.f;
  for (int m = tid; m < NN; m += 256) {
    float e = __expf(row[m] - rmax);
    row[m] = e; sum += e;
  }
  red[tid] = sum; __syncthreads();
  for (int o = 128; o > 0; o >>= 1) { if (tid < o) red[tid] += red[tid + o]; __syncthreads(); }
  float inv = 1.0f / red[0];
  for (int m = tid; m < NP; m += 256) {
    float v = (m < NN) ? row[m] * inv : 0.f;
    Abf[(long long)n * NP + m] = f2bf(v);
  }
}

// ---------------- x (B,T,N) f32 -> (T, NP, B) bf16 ----------------
__global__ void k_xpre(const float* __restrict__ x, unsigned short* __restrict__ xbf) {
  int idx = blockIdx.x * 256 + threadIdx.x;   // TIN*NP*BB
  if (idx >= TIN * NP * BB) return;
  int b = idx & (BB - 1);
  int n = (idx >> 5) & (NP - 1);
  int t = idx >> 16;
  float v = (n < NN) ? x[((long long)b * TIN + t) * NN + n] : 0.f;
  xbf[idx] = f2bf(v);
}

// ---------------- NAPL: per-node weights transposed (n, O, KD) bf16 + bias f32 ----------------
__global__ void k_napl(const float* __restrict__ Wp, const float* __restrict__ bp,
                       const float* __restrict__ emb,
                       unsigned short* __restrict__ WT, float* __restrict__ bn,
                       int Ireal, int O, int CP) {
  __shared__ float e[DD];
  int n = blockIdx.x;            // < NP
  int tid = threadIdx.x;
  if (tid < DD) e[tid] = (n < NN) ? emb[n * DD + tid] : 0.f;
  __syncthreads();
  int KD = 2 * CP;
  long long base = (long long)n * O * KD;
  long long dstride = (long long)2 * Ireal * O;
  for (int idx = tid; idx < O * KD; idx += 256) {
    int o = idx / KD;
    int ii = idx - o * KD;
    int k = ii / CP;
    int i = ii - k * CP;
    float v = 0.f;
    if (i < Ireal && n < NN) {
      const float* w = Wp + ((long long)k * Ireal + i) * O + o;
      #pragma unroll
      for (int d = 0; d < DD; ++d) v += e[d] * w[d * dstride];
    }
    WT[base + idx] = f2bf(v);
  }
  if (tid < O && n < NN) {
    float v = 0.f;
    #pragma unroll
    for (int d = 0; d < DD; ++d) v += e[d] * bp[d * O + tid];
    bn[(long long)n * O + tid] = v;
  }
}

// ---------------- build concat layer0: [x_t, (r*)h0] -> Act (n,b,c) + XT (col,n), bf16 ----------------
__global__ void k_build0(const unsigned short* __restrict__ xt, const float* __restrict__ h,
                         const float* __restrict__ ZR, int use_r,
                         unsigned short* __restrict__ Act, unsigned short* __restrict__ XT) {
  int idx = blockIdx.x * 256 + threadIdx.x;   // NP*BB
  int n = idx & (NP - 1);
  int b = idx >> 11;
  float xv = bf2f(xt[n * BB + b]);
  const float* hp = h + ((long long)n * BB + b) * HH;
  const float* rp = ZR + ((long long)n * BB + b) * (2 * HH) + HH;
  long long abase = ((long long)n * BB + b) * KD0;
  unsigned pack[4];
  for (int c0 = 0; c0 < CP0; c0 += 8) {
    #pragma unroll
    for (int j = 0; j < 8; ++j) {
      int c = c0 + j;
      float v;
      if (c == 0) v = xv;
      else if (c < 1 + HH) { float hv = hp[c - 1]; if (use_r) hv *= rp[c - 1]; v = hv; }
      else v = 0.f;
      unsigned short bv = f2bf(v);
      if ((j & 1) == 0) pack[j >> 1] = bv;
      else pack[j >> 1] |= ((unsigned)bv) << 16;
      XT[((long long)(b * CP0 + c)) * NP + n] = bv;
    }
    *reinterpret_cast<uint4*>(&((unsigned short*)Act)[abase + c0]) =
        *reinterpret_cast<const uint4*>(pack);
  }
}

// ---------------- build concat layer1: [h0, (r*)h1] ----------------
__global__ void k_build1(const float* __restrict__ h0, const float* __restrict__ h1,
                         const float* __restrict__ ZR, int use_r,
                         unsigned short* __restrict__ Act, unsigned short* __restrict__ XT) {
  int idx = blockIdx.x * 256 + threadIdx.x;   // NP*BB
  int n = idx & (NP - 1);
  int b = idx >> 11;
  const float* p0 = h0 + ((long long)n * BB + b) * HH;
  const float* p1 = h1 + ((long long)n * BB + b) * HH;
  const float* rp = ZR + ((long long)n * BB + b) * (2 * HH) + HH;
  long long abase = ((long long)n * BB + b) * KD1;
  unsigned pack[4];
  for (int c0 = 0; c0 < CP1; c0 += 8) {
    #pragma unroll
    for (int j = 0; j < 8; ++j) {
      int c = c0 + j;
      float v;
      if (c < HH) v = p0[c];
      else { float hv = p1[c - HH]; if (use_r) hv *= rp[c - HH]; v = hv; }
      unsigned short bv = f2bf(v);
      if ((j & 1) == 0) pack[j >> 1] = bv;
      else pack[j >> 1] |= ((unsigned)bv) << 16;
      XT[((long long)(b * CP1 + c)) * NP + n] = bv;
    }
    *reinterpret_cast<uint4*>(&((unsigned short*)Act)[abase + c0]) =
        *reinterpret_cast<const uint4*>(pack);
  }
}

// ---- WMMA fragment loaders (ISA VGPR layouts) ----
__device__ __forceinline__ v16bf ldfragA(const unsigned short* p) {
  // A: per lane two contiguous 8-elem chunks at +0 and +16 (K halves)
  v16bf a;
  uint4* av = reinterpret_cast<uint4*>(&a);
  av[0] = *reinterpret_cast<const uint4*>(p);
  av[1] = *reinterpret_cast<const uint4*>(p + 16);
  return a;
}
__device__ __forceinline__ v16bf ldfragB(const unsigned short* p) {
  // B (via transposed storage): per lane two contiguous 8-elem chunks at +0 and +16 (K halves)
  v16bf b;
  uint4* bv = reinterpret_cast<uint4*>(&b);
  bv[0] = *reinterpret_cast<const uint4*>(p);
  bv[1] = *reinterpret_cast<const uint4*>(p + 16);
  return b;
}

// ---------------- GEMM1: Y = A(NPxNP) @ X(NP x B*CP); 2x2 register blocking ----------------
template <int CP>
__global__ void k_gemm1(const unsigned short* __restrict__ Abf,
                        const unsigned short* __restrict__ XT,
                        unsigned short* __restrict__ Act) {
  const int KD = 2 * CP;
  int wave = threadIdx.x >> 5;
  int lane = threadIdx.x & 31;
  int hi   = lane >> 4;
  int lrow = lane & 15;
  int m0    = blockIdx.x * 32;                 // 2 row tiles
  int nbase = blockIdx.y * 256 + wave * 32;    // 2 col tiles per wave
  const unsigned short* a0p = Abf + (long long)(m0 + lrow) * NP + hi * 8;
  const unsigned short* a1p = a0p + (long long)16 * NP;
  const unsigned short* b0p = XT + (long long)(nbase + lrow) * NP + hi * 8;
  const unsigned short* b1p = b0p + (long long)16 * NP;
  v8f acc00 = {}, acc01 = {}, acc10 = {}, acc11 = {};
  for (int kk = 0; kk < NP; kk += 32) {
    v16bf a0 = ldfragA(a0p + kk);
    v16bf a1 = ldfragA(a1p + kk);
    v16bf b0 = ldfragB(b0p + kk);
    v16bf b1 = ldfragB(b1p + kk);
    acc00 = __builtin_amdgcn_wmma_f32_16x16x32_bf16(false, a0, false, b0, (short)0, acc00, false, false);
    acc01 = __builtin_amdgcn_wmma_f32_16x16x32_bf16(false, a0, false, b1, (short)0, acc01, false, false);
    acc10 = __builtin_amdgcn_wmma_f32_16x16x32_bf16(false, a1, false, b0, (short)0, acc10, false, false);
    acc11 = __builtin_amdgcn_wmma_f32_16x16x32_bf16(false, a1, false, b1, (short)0, acc11, false, false);
  }
  auto store_tile = [&](const v8f& acc, int mb, int col) {
    int bidx = col / CP;
    int c    = col - bidx * CP;
    #pragma unroll
    for (int r = 0; r < 8; ++r) {
      int m = mb + hi * 8 + r;
      Act[((long long)m * BB + bidx) * KD + CP + c] = f2bf(acc[r]);
    }
  };
  store_tile(acc00, m0,      nbase + lrow);
  store_tile(acc01, m0,      nbase + 16 + lrow);
  store_tile(acc10, m0 + 16, nbase + lrow);
  store_tile(acc11, m0 + 16, nbase + 16 + lrow);
}

// ---------------- GEMM2: per-node (B x KD) @ (KD x O), TDM-staged LDS, fused epilogue ----------------
// MODE 0: gate  -> ZR = sigmoid(acc + bias)            (O = 2H)
// MODE 1: cand  -> h  = z*h + (1-z)*tanh(acc + bias)   (O = H)
template <int KD, int O, int MODE>
__global__ void k_gemm2(const unsigned short* __restrict__ Act,
                        const unsigned short* __restrict__ WT,
                        const float* __restrict__ bias,
                        float* __restrict__ ZR,
                        const float* __restrict__ hin,
                        float* __restrict__ hout) {
  __shared__ unsigned short Ash[32 * KD];
  int n = blockIdx.x;                     // < NN
  int tid = threadIdx.x;
  // ---- Tensor Data Mover: DMA this node's 32xKD bf16 tile (as dwords) into LDS ----
  if (tid < 32) {                         // wave 0 issues the descriptor
    unsigned ldsoff = (unsigned)(unsigned long long)(void*)&Ash[0];  // LDS offset = low 32 bits of address
    unsigned long long ga = (unsigned long long)(const void*)(Act + (long long)n * BB * KD);
    const unsigned ELEMS = (unsigned)(32 * KD) / 2;  // dword elements
    u32x4 g0;
    g0[0] = 1u;                                        // count=1, user mode
    g0[1] = ldsoff;                                    // lds_addr
    g0[2] = (unsigned)(ga & 0xffffffffu);              // global_addr[31:0]
    g0[3] = (unsigned)((ga >> 32) & 0x01ffffffu) | (2u << 30);  // global_addr[56:32], type=2
    i32x8 g1;
    g1[0] = 0x20000;                                   // data_size = 4B
    g1[1] = (int)(ELEMS << 16);                        // tensor_dim0[15:0] @ bits 63:48
    g1[2] = (int)((ELEMS >> 16) | (1u << 16));         // tensor_dim0[31:16], tensor_dim1=1
    g1[3] = (int)(ELEMS << 16);                        // tile_dim0 @ bits 127:112
    g1[4] = 1;                                         // tile_dim1 = 1
    g1[5] = (int)ELEMS;                                // tensor_dim0_stride[31:0]
    g1[6] = 0;
    g1[7] = 0;
    i32x4 gz4 = {0, 0, 0, 0};
    i32x8 gz8 = {0, 0, 0, 0, 0, 0, 0, 0};
    __builtin_amdgcn_tensor_load_to_lds(g0, g1, gz4, gz4, gz8, 0);
    __builtin_amdgcn_s_wait_tensorcnt(0);
  }
  __syncthreads();
  int wave = tid >> 5, lane = tid & 31, hi = lane >> 4, lrow = lane & 15;
  const unsigned short* ap0 = Ash + lrow * KD + hi * 8;
  const unsigned short* ap1 = Ash + (16 + lrow) * KD + hi * 8;
  for (int ot = wave; ot < O / 16; ot += 8) {
    int ocol = ot * 16 + lrow;
    v8f acc0 = {}, acc1 = {};
    const unsigned short* wp = WT + ((long long)n * O + ocol) * KD + hi * 8;
    #pragma unroll
    for (int kk = 0; kk < KD; kk += 32) {
      v16bf b  = ldfragB(wp + kk);
      v16bf a0 = ldfragA(ap0 + kk);   // ds_load_b128 x2
      v16bf a1 = ldfragA(ap1 + kk);
      acc0 = __builtin_amdgcn_wmma_f32_16x16x32_bf16(false, a0, false, b, (short)0, acc0, false, false);
      acc1 = __builtin_amdgcn_wmma_f32_16x16x32_bf16(false, a1, false, b, (short)0, acc1, false, false);
    }
    float bo = bias[(long long)n * O + ocol];
    #pragma unroll
    for (int r = 0; r < 8; ++r) {
      int b0 = hi * 8 + r;         // batch rows for m-tile 0
      int b1 = 16 + hi * 8 + r;    // batch rows for m-tile 1
      float v0 = acc0[r] + bo;
      float v1 = acc1[r] + bo;
      if (MODE == 0) {
        ZR[((long long)n * BB + b0) * (2 * HH) + ocol] = 1.0f / (1.0f + __expf(-v0));
        ZR[((long long)n * BB + b1) * (2 * HH) + ocol] = 1.0f / (1.0f + __expf(-v1));
      } else {
        float z0 = ZR[((long long)n * BB + b0) * (2 * HH) + ocol];
        float z1 = ZR[((long long)n * BB + b1) * (2 * HH) + ocol];
        float h0v = hin[((long long)n * BB + b0) * HH + ocol];
        float h1v = hin[((long long)n * BB + b1) * HH + ocol];
        hout[((long long)n * BB + b0) * HH + ocol] = z0 * h0v + (1.0f - z0) * tanhf(v0);
        hout[((long long)n * BB + b1) * HH + ocol] = z1 * h1v + (1.0f - z1) * tanhf(v1);
      }
    }
  }
}

// ---------------- head: y[b,o,n] = h1[n,b,:] . head_w[o,:] + head_b[o] ----------------
__global__ void k_head(const float* __restrict__ h1, const float* __restrict__ hw,
                       const float* __restrict__ hb, float* __restrict__ out) {
  int idx = blockIdx.x * 256 + threadIdx.x;
  if (idx >= BB * TOUT * NN) return;
  int n = idx % NN;
  int o = (idx / NN) % TOUT;
  int b = idx / (NN * TOUT);
  const float* hp = h1 + ((long long)n * BB + b) * HH;
  float s = hb[o];
  #pragma unroll
  for (int k = 0; k < HH; ++k) s += hp[k] * hw[o * HH + k];
  out[idx] = s;
}

// =====================================================================

extern "C" void kernel_launch(void* const* d_in, const int* in_sizes, int n_in,
                              void* d_out, int out_size, void* d_ws, size_t ws_size,
                              hipStream_t stream) {
  (void)in_sizes; (void)n_in; (void)out_size; (void)ws_size;

  const float* x         = (const float*)d_in[0];
  const float* node_emb  = (const float*)d_in[1];
  const float* adapt_emb = (const float*)d_in[2];
  const float* Wg0 = (const float*)d_in[3];
  const float* bg0 = (const float*)d_in[4];
  const float* Wc0 = (const float*)d_in[5];
  const float* bc0 = (const float*)d_in[6];
  const float* Wg1 = (const float*)d_in[7];
  const float* bg1 = (const float*)d_in[8];
  const float* Wc1 = (const float*)d_in[9];
  const float* bc1 = (const float*)d_in[10];
  const float* head_w = (const float*)d_in[11];
  const float* head_b = (const float*)d_in[12];
  float* out = (float*)d_out;

  // ---- workspace carve-up ----
  char* ws = (char*)d_ws;
  size_t off = 0;
  auto alloc = [&](size_t bytes) { char* p = ws + off; off += (bytes + 255) & ~(size_t)255; return p; };

  unsigned short* A_bf = (unsigned short*)alloc((size_t)NP * NP * 2);
  unsigned short* xbf  = (unsigned short*)alloc((size_t)TIN * NP * BB * 2);
  float* h0 = (float*)alloc((size_t)NP * BB * HH * 4);
  float* h1 = (float*)alloc((size_t)NP * BB * HH * 4);
  float* ZR = (float*)alloc((size_t)NP * BB * 2 * HH * 4);
  unsigned short* Act0 = (unsigned short*)alloc((size_t)NP * BB * KD0 * 2);
  unsigned short* XT0  = (unsigned short*)alloc((size_t)BC0 * NP * 2);
  unsigned short* Act1 = (unsigned short*)alloc((size_t)NP * BB * KD1 * 2);
  unsigned short* XT1  = (unsigned short*)alloc((size_t)BC1 * NP * 2);
  unsigned short* WgT0 = (unsigned short*)alloc((size_t)NP * 128 * KD0 * 2);
  float* bg0n = (float*)alloc((size_t)NP * 128 * 4);
  unsigned short* WcT0 = (unsigned short*)alloc((size_t)NP * 64 * KD0 * 2);
  float* bc0n = (float*)alloc((size_t)NP * 64 * 4);
  unsigned short* WgT1 = (unsigned short*)alloc((size_t)NP * 128 * KD1 * 2);
  float* bg1n = (float*)alloc((size_t)NP * 128 * 4);
  unsigned short* WcT1 = (unsigned short*)alloc((size_t)NP * 64 * KD1 * 2);
  float* bc1n = (float*)alloc((size_t)NP * 64 * 4);

  // ---- init (deterministic every call) ----
  k_zero<<<1024, 256, 0, stream>>>((unsigned int*)A_bf, (long long)NP * NP / 2);
  k_zero<<<1024, 256, 0, stream>>>((unsigned int*)h0, (long long)NP * BB * HH);
  k_zero<<<1024, 256, 0, stream>>>((unsigned int*)h1, (long long)NP * BB * HH);
  k_zero<<<1024, 256, 0, stream>>>((unsigned int*)ZR, (long long)NP * BB * 2 * HH);

  k_adj<<<NN, 256, 0, stream>>>(adapt_emb, A_bf);
  k_xpre<<<(TIN * NP * BB) / 256, 256, 0, stream>>>(x, xbf);

  k_napl<<<NP, 256, 0, stream>>>(Wg0, bg0, node_emb, WgT0, bg0n, 1 + HH, 2 * HH, CP0);
  k_napl<<<NP, 256, 0, stream>>>(Wc0, bc0, node_emb, WcT0, bc0n, 1 + HH, HH, CP0);
  k_napl<<<NP, 256, 0, stream>>>(Wg1, bg1, node_emb, WgT1, bg1n, 2 * HH, 2 * HH, CP1);
  k_napl<<<NP, 256, 0, stream>>>(Wc1, bc1, node_emb, WcT1, bc1n, 2 * HH, HH, CP1);

  const dim3 g1a(NP / 32, BC0 / 256);   // 64 x 12 blocks, 8 waves, 32x32 per wave
  const dim3 g1b(NP / 32, BC1 / 256);   // 64 x 16

  for (int t = 0; t < TIN; ++t) {
    const unsigned short* xt = xbf + (size_t)t * NP * BB;
    // ---- cell 0 ----
    k_build0<<<(NP * BB) / 256, 256, 0, stream>>>(xt, h0, ZR, 0, Act0, XT0);
    k_gemm1<CP0><<<g1a, 256, 0, stream>>>(A_bf, XT0, Act0);
    k_gemm2<KD0, 2 * HH, 0><<<NN, 256, 0, stream>>>(Act0, WgT0, bg0n, ZR, nullptr, nullptr);
    k_build0<<<(NP * BB) / 256, 256, 0, stream>>>(xt, h0, ZR, 1, Act0, XT0);
    k_gemm1<CP0><<<g1a, 256, 0, stream>>>(A_bf, XT0, Act0);
    k_gemm2<KD0, HH, 1><<<NN, 256, 0, stream>>>(Act0, WcT0, bc0n, ZR, h0, h0);
    // ---- cell 1 ----
    k_build1<<<(NP * BB) / 256, 256, 0, stream>>>(h0, h1, ZR, 0, Act1, XT1);
    k_gemm1<CP1><<<g1b, 256, 0, stream>>>(A_bf, XT1, Act1);
    k_gemm2<KD1, 2 * HH, 0><<<NN, 256, 0, stream>>>(Act1, WgT1, bg1n, ZR, nullptr, nullptr);
    k_build1<<<(NP * BB) / 256, 256, 0, stream>>>(h0, h1, ZR, 1, Act1, XT1);
    k_gemm1<CP1><<<g1b, 256, 0, stream>>>(A_bf, XT1, Act1);
    k_gemm2<KD1, HH, 1><<<NN, 256, 0, stream>>>(Act1, WcT1, bc1n, ZR, h1, h1);
  }

  k_head<<<(BB * TOUT * NN + 255) / 256, 256, 0, stream>>>(h1, head_w, head_b, out);
}
